// GCN_60284160966674
// MI455X (gfx1250) — matrix operand order repaired and checked by reading, not compiled, not run on hardware
//
#include <hip/hip_runtime.h>
#include <hip/hip_bf16.h>

// ---------------------------------------------------------------------------
// GCN 3-layer forward on gfx1250 (MI455X), wave32.
//   - GEMMs: v_wmma_f32_16x16x32_bf16, A panel in VGPRs, W transposed bf16 in
//     LDS (ds_load_b128 fragments). M templated -> immediate-offset stores.
//   - Aggregation: float4-vectorized coalesced global_atomic_add_f32 scatter.
// Working set (~90 MB) is L2-resident (192 MB); scatter atomics dominate.
// ---------------------------------------------------------------------------

#define N_NODES 50000
#define N_EDGES 640000
#define F_IN    128
#define HID     128
#define NCLS    40

typedef __attribute__((ext_vector_type(16))) __bf16 v16bf;
typedef __attribute__((ext_vector_type(8)))  float  v8f;

// -------------------------------- utility kernels --------------------------

__global__ void fill_f32_v4(float4* __restrict__ p, float v, long long n4) {
    long long i = (long long)blockIdx.x * blockDim.x + threadIdx.x;
    if (i < n4) p[i] = make_float4(v, v, v, v);
}

// deg[dst] += w  (deg pre-initialized to 1.0 == self-loop weight)
__global__ void deg_kernel(float* __restrict__ deg,
                           const int* __restrict__ dst,
                           const float* __restrict__ w, int e) {
    int i = blockIdx.x * blockDim.x + threadIdx.x;
    if (i < e) atomicAdd(&deg[dst[i]], w[i]);
}

// deg -> dinv (deg >= 1.0 always, so no zero guard needed)
__global__ void rsqrt_kernel(float* __restrict__ deg, int n) {
    int i = blockIdx.x * blockDim.x + threadIdx.x;
    if (i < n) deg[i] = rsqrtf(deg[i]);
}

__global__ void norm_kernel(float* __restrict__ norm,
                            const float* __restrict__ dinv,
                            const int* __restrict__ src,
                            const int* __restrict__ dst,
                            const float* __restrict__ w, int e) {
    int i = blockIdx.x * blockDim.x + threadIdx.x;
    if (i < e) norm[i] = dinv[src[i]] * w[i] * dinv[dst[i]];
}

// out[dst[e]][j..j+3] += h[src[e]][j..j+3] * norm[e]
// One thread = one float4: consecutive threads share an edge and walk j, so
// loads are b128-coalesced and the 4 atomics of each thread hit one cacheline.
template <int M>
__global__ void scatter_kernel(float* __restrict__ out,
                               const float* __restrict__ h,
                               const float* __restrict__ norm,
                               const int* __restrict__ src,
                               const int* __restrict__ dst,
                               long long total4) {
    constexpr int M4 = M / 4;
    long long idx = (long long)blockIdx.x * blockDim.x + threadIdx.x;
    if (idx >= total4) return;
    int e = (int)(idx / M4);
    int j = (int)(idx - (long long)e * M4) * 4;
    float nv = norm[e];
    const float4 hv = *(const float4*)(h + (long long)src[e] * M + j);
    float* o = out + (long long)dst[e] * M + j;
    atomicAdd(o + 0, hv.x * nv);
    atomicAdd(o + 1, hv.y * nv);
    atomicAdd(o + 2, hv.z * nv);
    atomicAdd(o + 3, hv.w * nv);
}

// buf[i][j] = act(buf[i][j] + h[i][j]*dinv[i]^2 + bias[j])  (self-loop + bias)
template <int M>
__global__ void fuse_kernel(float* __restrict__ buf,
                            const float* __restrict__ h,
                            const float* __restrict__ dinv,
                            const float* __restrict__ bias,
                            long long total4, int relu) {
    constexpr int M4 = M / 4;
    long long idx = (long long)blockIdx.x * blockDim.x + threadIdx.x;
    if (idx >= total4) return;
    int i = (int)(idx / M4);
    int j = (int)(idx - (long long)i * M4) * 4;
    float d = dinv[i];
    d = d * d;
    float4 a  = *(const float4*)(buf + (long long)i * M + j);
    float4 hv = *(const float4*)(h + (long long)i * M + j);
    float4 bs = *(const float4*)(bias + j);
    float4 r;
    r.x = fmaf(hv.x, d, a.x) + bs.x;
    r.y = fmaf(hv.y, d, a.y) + bs.y;
    r.z = fmaf(hv.z, d, a.z) + bs.z;
    r.w = fmaf(hv.w, d, a.w) + bs.w;
    if (relu) {
        r.x = fmaxf(r.x, 0.0f); r.y = fmaxf(r.y, 0.0f);
        r.z = fmaxf(r.z, 0.0f); r.w = fmaxf(r.w, 0.0f);
    }
    *(float4*)(buf + (long long)i * M + j) = r;
}

// -------------------------------- WMMA GEMM --------------------------------
// out[N x M] = A[N x 128] (f32, cvt->bf16) @ W[128 x M] (f32, cvt->bf16 LDS)
// One wave per 16-row tile; 8 waves/block share W transposed in LDS.
// Layouts used here (wave32; A and B share a contiguous k order):
//   A frag:  lane lr holds row m0+lr; elems 0-15 = K lg*16+0..15
//   B frag:  lane lr holds col n0+lr; elems 0-15 = K lg*16+0..15 (contig in Wt)
//   C/D:     vgpr r -> row m0+lg*8+r, col n0+lr
template <int M, int MT>   // M = actual cols (128|40), MT = 16-col tiles (8|3)
__global__ void __launch_bounds__(256)
gemm_wmma_bf16(const float* __restrict__ A,
               const float* __restrict__ W,
               float* __restrict__ out,
               int rowtiles)   // 3125
{
    constexpr int MPAD = MT * 16;
    __shared__ __align__(32) __bf16 Wt[MPAD * 128];  // [n][k]

    const int tid = threadIdx.x;

    // Cooperative transpose+convert of W into LDS (zero-pad cols >= M).
    for (int idx = tid; idx < MPAD * 128; idx += 256) {
        int n = idx >> 7;          // column of W
        int k = idx & 127;         // row of W
        float v = (n < M) ? W[k * M + n] : 0.0f;
        Wt[n * 128 + k] = (__bf16)v;
    }
    __syncthreads();

    const int wave = tid >> 5;
    const int lane = tid & 31;
    const int lr   = lane & 15;
    const int lg   = lane >> 4;

    const int rt = blockIdx.x * 8 + wave;
    if (rt >= rowtiles) return;
    const int m0 = rt * 16;

    // Load this wave's 16x128 A panel as 4 bf16 fragments (K chunks of 32).
    const float* Arow = A + (long long)(m0 + lr) * 128;
    v16bf afrag[4];
#pragma unroll
    for (int kc = 0; kc < 4; ++kc) {
        v16bf av;
        const int kA = kc * 32 + lg * 16;
#pragma unroll
        for (int j = 0; j < 8; ++j) av[j]     = (__bf16)Arow[kA + j];
#pragma unroll
        for (int j = 0; j < 8; ++j) av[8 + j] = (__bf16)Arow[kA + 8 + j];
        afrag[kc] = av;
    }

#pragma unroll
    for (int ct = 0; ct < MT; ++ct) {
        const int n0 = ct * 16;
        v8f acc = {};
#pragma unroll
        for (int kc = 0; kc < 4; ++kc) {
            const int kb = kc * 32 + lg * 16;
            v16bf bfrag = *reinterpret_cast<const v16bf*>(
                &Wt[(n0 + lr) * 128 + kb]);
            acc = __builtin_amdgcn_wmma_f32_16x16x32_bf16(
                false, afrag[kc], false, bfrag, (short)0, acc, false, false);
        }
        const int col = n0 + lr;
        if (col < M) {   // compile-time true except last tile when M==40
            float* orow = out + (long long)(m0 + lg * 8) * M + col;
#pragma unroll
            for (int r = 0; r < 8; ++r)
                orow[r * M] = acc[r];   // immediate offsets (M compile-time)
        }
    }
}

// -------------------------------- launcher ---------------------------------

static inline int ceil_div(long long a, int b) { return (int)((a + b - 1) / b); }

extern "C" void kernel_launch(void* const* d_in, const int* in_sizes, int n_in,
                              void* d_out, int out_size, void* d_ws, size_t ws_size,
                              hipStream_t stream) {
    (void)in_sizes; (void)n_in; (void)out_size; (void)ws_size;

    const float* x  = (const float*)d_in[0];
    const int*   ei = (const int*)d_in[1];      // (2, E): src row then dst row
    const float* ew = (const float*)d_in[2];
    const float* W1 = (const float*)d_in[3];
    const float* b1 = (const float*)d_in[4];
    const float* W2 = (const float*)d_in[5];
    const float* b2 = (const float*)d_in[6];
    const float* W3 = (const float*)d_in[7];
    const float* b3 = (const float*)d_in[8];
    float* out = (float*)d_out;

    const int* src = ei;
    const int* dst = ei + N_EDGES;

    // Workspace carve-up (floats): deg | norm | bufA | bufB | bufC  (~62 MB)
    float* deg  = (float*)d_ws;                    // N
    float* norm = deg  + N_NODES;                  // E
    float* bufA = norm + N_EDGES;                  // N*128 (GEMM out / ping)
    float* bufB = bufA + (long long)N_NODES * HID; // N*128 (agg out / pong)
    float* bufC = bufB + (long long)N_NODES * HID; // N*40  (layer-3 GEMM out)

    const int T = 256;
    const long long NH4 = (long long)N_NODES * HID / 4;
    const long long NC4 = (long long)N_NODES * NCLS / 4;
    const long long EH4 = (long long)N_EDGES * HID / 4;
    const long long EC4 = (long long)N_EDGES * NCLS / 4;
    const int rowtiles = N_NODES / 16;             // 3125 exactly
    const int gemm_blocks = (rowtiles + 7) / 8;

    // ---- normalization: deg -> dinv -> norm --------------------------------
    fill_f32_v4<<<ceil_div(N_NODES / 4, T), T, 0, stream>>>((float4*)deg, 1.0f, N_NODES / 4);
    deg_kernel<<<ceil_div(N_EDGES, T), T, 0, stream>>>(deg, dst, ew, N_EDGES);
    rsqrt_kernel<<<ceil_div(N_NODES, T), T, 0, stream>>>(deg, N_NODES);
    norm_kernel<<<ceil_div(N_EDGES, T), T, 0, stream>>>(norm, deg, src, dst, ew, N_EDGES);

    // ---- layer 1: x @ W1 -> bufA; aggregate -> bufB; +b1, ReLU -------------
    gemm_wmma_bf16<HID, 8><<<gemm_blocks, T, 0, stream>>>(x, W1, bufA, rowtiles);
    fill_f32_v4<<<ceil_div(NH4, T), T, 0, stream>>>((float4*)bufB, 0.0f, NH4);
    scatter_kernel<HID><<<ceil_div(EH4, T), T, 0, stream>>>(bufB, bufA, norm, src, dst, EH4);
    fuse_kernel<HID><<<ceil_div(NH4, T), T, 0, stream>>>(bufB, bufA, deg, b1, NH4, 1);

    // ---- layer 2: bufB @ W2 -> bufA; aggregate -> bufB; +b2, ReLU ----------
    gemm_wmma_bf16<HID, 8><<<gemm_blocks, T, 0, stream>>>(bufB, W2, bufA, rowtiles);
    fill_f32_v4<<<ceil_div(NH4, T), T, 0, stream>>>((float4*)bufB, 0.0f, NH4);
    scatter_kernel<HID><<<ceil_div(EH4, T), T, 0, stream>>>(bufB, bufA, norm, src, dst, EH4);
    fuse_kernel<HID><<<ceil_div(NH4, T), T, 0, stream>>>(bufB, bufA, deg, b2, NH4, 1);

    // ---- layer 3: bufB @ W3 -> bufC; aggregate -> out; +b3 (no ReLU) -------
    gemm_wmma_bf16<NCLS, 3><<<gemm_blocks, T, 0, stream>>>(bufB, W3, bufC, rowtiles);
    fill_f32_v4<<<ceil_div(NC4, T), T, 0, stream>>>((float4*)out, 0.0f, NC4);
    scatter_kernel<NCLS><<<ceil_div(EC4, T), T, 0, stream>>>(out, bufC, norm, src, dst, EC4);
    fuse_kernel<NCLS><<<ceil_div(NC4, T), T, 0, stream>>>(out, bufC, deg, b3, NC4, 0);
}
